// EGATNet_41721312313872
// MI455X (gfx1250) — matrix operand order repaired and checked by reading, not compiled, not run on hardware
//
#include <hip/hip_runtime.h>

#define N_NODES 50000
#define N_EDGES 600000
#define HEADS 3
#define FDIM 64
#define HD 192
#define N_G 128
#define N_C 10
#define BN_BLOCKS 256

typedef __attribute__((ext_vector_type(16))) __bf16 v16bf;
typedef __attribute__((ext_vector_type(8)))  float  v8f;

// ---------- helpers ----------
__device__ __forceinline__ unsigned fenc(float f) {
  unsigned u = __float_as_uint(f);
  return (u & 0x80000000u) ? ~u : (u | 0x80000000u);
}
__device__ __forceinline__ float fdec(unsigned u) {
  u = (u & 0x80000000u) ? (u & 0x7fffffffu) : ~u;
  return __uint_as_float(u);
}
__device__ __forceinline__ float half16_reduce(float v) {
  v += __shfl_xor(v, 1, 32);
  v += __shfl_xor(v, 2, 32);
  v += __shfl_xor(v, 4, 32);
  v += __shfl_xor(v, 8, 32);
  return v;  // sum over the 16 lanes of each half-wave
}

__global__ __launch_bounds__(256) void fill_zero(float* __restrict__ p, long long n) {
  long long i = (long long)blockIdx.x * 256 + threadIdx.x;
  if (i < n) p[i] = 0.f;
}

// ---------- fused WMMA GEMM + attention-dot epilogue ----------
// Y[M,192] = act(X[M,Kin]) @ W[Kin,192]   (bf16 WMMA, f32 accumulate)
// MODE 0 (node): write Yfull=h[M,192], Yred=ds[M,3]=h.a_s, Yred2=dd[M,3]=h.a_d
// MODE 1 (edge): write Yred=eo[M,64]=mean_heads(eh), Yred2=de[M,3]=eh.a_e
template<int MODE>
__global__ __launch_bounds__(128)
void gemm_att(const float* __restrict__ X, const float* __restrict__ W,
              const float* __restrict__ av0, const float* __restrict__ av1,
              float* __restrict__ Yfull, float* __restrict__ Yred,
              float* __restrict__ Yred2, int M, int Kin, int relu_in)
{
  __shared__ __bf16 Wt[HD * 64];   // W transposed: Wt[col*64 + k], K zero-padded to 64
  __shared__ __bf16 At[64 * 64];   // A tile: At[row*64 + k]
  const int tid  = threadIdx.x;
  const int wave = tid >> 5, lane = tid & 31;
  const int half = lane >> 4, ln = lane & 15;
  const int row0 = blockIdx.x * 64;
  const bool fullTile = (row0 + 64 <= M);   // uniform: all blocks except (maybe) the last

  for (int i = tid; i < HD * 64; i += 128) {
    int c = i >> 6, k = i & 63;
    Wt[i] = (__bf16)((k < Kin) ? W[(size_t)k * HD + c] : 0.f);
  }
  for (int i = tid; i < 64 * 64; i += 128) {
    int r = i >> 6, k = i & 63;
    int gr = row0 + r;
    float v = 0.f;
    if (gr < M && k < Kin) {
      v = X[(size_t)gr * Kin + k];
      if (relu_in) v = fmaxf(v, 0.f);
    }
    At[i] = (__bf16)v;
  }
  __syncthreads();

  // A fragments for K-steps 0..31 and 32..63 (16-bit A 16x32 layout, ISA 7.12.2)
  const int mrow = wave * 16 + ln;
  v16bf a0, a1;
#pragma unroll
  for (int vg = 0; vg < 8; ++vg) {
    int kb = ((vg < 4) ? (2 * vg) : (16 + 2 * (vg - 4))) + 8 * half;
    a0[2*vg]   = At[mrow * 64 + kb];
    a0[2*vg+1] = At[mrow * 64 + kb + 1];
    a1[2*vg]   = At[mrow * 64 + 32 + kb];
    a1[2*vg+1] = At[mrow * 64 + 32 + kb + 1];
  }

  // per-lane base row for this wave's D tile rows (r in 0..7 -> row rbase + r)
  const int rbase = row0 + wave * 16 + 8 * half;
  float* __restrict__ yp = (MODE == 0) ? (Yfull + (size_t)rbase * HD + ln) : nullptr;
  float* __restrict__ ep = (MODE == 1) ? (Yred  + (size_t)rbase * 64 + ln) : nullptr;

  float ps[8][3], pd2[8][3], pe[8][4];
#pragma unroll
  for (int r = 0; r < 8; ++r) {
#pragma unroll
    for (int hh = 0; hh < 3; ++hh) { ps[r][hh] = 0.f; pd2[r][hh] = 0.f; }
#pragma unroll
    for (int q = 0; q < 4; ++q) pe[r][q] = 0.f;
  }

#pragma unroll
  for (int j = 0; j < 12; ++j) {       // 12 column tiles of 16 -> 192 outputs
    v8f acc = {0.f,0.f,0.f,0.f,0.f,0.f,0.f,0.f};
    const int cgl = 16 * j + ln;       // global output column for this lane
    v16bf b;
#pragma unroll
    for (int vg = 0; vg < 8; ++vg) {   // B fragment, kstep 0 (K-pairs contiguous in Wt)
      int K = ((vg < 4) ? (2 * vg) : (16 + 2 * (vg - 4))) + 8 * half;
      b[2*vg]   = Wt[cgl * 64 + K];
      b[2*vg+1] = Wt[cgl * 64 + K + 1];
    }
    acc = __builtin_amdgcn_wmma_f32_16x16x32_bf16(false, a0, false, b, (short)0, acc, false, false);
#pragma unroll
    for (int vg = 0; vg < 8; ++vg) {   // B fragment, kstep 1
      int K = 32 + ((vg < 4) ? (2 * vg) : (16 + 2 * (vg - 4))) + 8 * half;
      b[2*vg]   = Wt[cgl * 64 + K];
      b[2*vg+1] = Wt[cgl * 64 + K + 1];
    }
    acc = __builtin_amdgcn_wmma_f32_16x16x32_bf16(false, a1, false, b, (short)0, acc, false, false);

    const int head = j >> 2;
    if constexpr (MODE == 0) {
      // h store: base pointer + immediate offsets; unguarded on full tiles
      if (fullTile) {
#pragma unroll
        for (int r = 0; r < 8; ++r) yp[(size_t)r * HD + 16 * j] = acc[r];
      } else {
#pragma unroll
        for (int r = 0; r < 8; ++r)
          if (rbase + r < M) yp[(size_t)r * HD + 16 * j] = acc[r];
      }
      const float avs = av0[cgl];      // a_s[head, d] : head*64+d == cgl
      const float avd = av1[cgl];      // a_d[head, d]
#pragma unroll
      for (int r = 0; r < 8; ++r) {
        ps[r][head]  += acc[r] * avs;
        pd2[r][head] += acc[r] * avd;
      }
    } else {
      const float ave = av0[cgl];      // a_e[head, d]
#pragma unroll
      for (int r = 0; r < 8; ++r) {
        ps[r][head] += acc[r] * ave;
        pe[r][j & 3] += acc[r];        // head-sum for eo
      }
    }
  }

  // ---- epilogue: eo direct store, ds/dd/de via half-wave reductions ----
  if constexpr (MODE == 1) {
    if (fullTile) {
#pragma unroll
      for (int r = 0; r < 8; ++r)
#pragma unroll
        for (int q = 0; q < 4; ++q)
          ep[(size_t)r * 64 + 16 * q] = pe[r][q] * (1.f / 3.f);
    } else {
#pragma unroll
      for (int r = 0; r < 8; ++r)
        if (rbase + r < M)
#pragma unroll
          for (int q = 0; q < 4; ++q)
            ep[(size_t)r * 64 + 16 * q] = pe[r][q] * (1.f / 3.f);
    }
  }
#pragma unroll
  for (int r = 0; r < 8; ++r) {
    const int m = rbase + r;
#pragma unroll
    for (int hh = 0; hh < 3; ++hh) {
      float vs = half16_reduce(ps[r][hh]);
      if constexpr (MODE == 0) {
        float vd = half16_reduce(pd2[r][hh]);
        if (ln == 0 && (fullTile || m < M)) {
          Yred [(size_t)m * 3 + hh] = vs;
          Yred2[(size_t)m * 3 + hh] = vd;
        }
      } else {
        if (ln == 0 && (fullTile || m < M)) Yred2[(size_t)m * 3 + hh] = vs;
      }
    }
  }
}

// ---------- attention pipeline ----------
__global__ __launch_bounds__(256)
void att_logits_max(const float* __restrict__ dsn, const float* __restrict__ ddn,
                    const float* __restrict__ dee, const int* __restrict__ src,
                    const int* __restrict__ dst, float* __restrict__ tb,
                    unsigned* __restrict__ menc, int E)
{
  int e = blockIdx.x * 256 + threadIdx.x;
  if (e >= E) return;
  int s = src[e], d = dst[e];
#pragma unroll
  for (int h = 0; h < 3; ++h) {
    float t = dsn[s*3+h] + ddn[d*3+h] + dee[(size_t)e*3+h];
    float lr = (t > 0.f) ? t : 0.2f * t;     // leaky_relu 0.2
    tb[(size_t)e*3+h] = lr;
    atomicMax(&menc[d*3+h], fenc(lr));
  }
}

__global__ __launch_bounds__(256)
void att_expsum(float* __restrict__ tb, const int* __restrict__ dst,
                const unsigned* __restrict__ menc, float* __restrict__ ssum, int E)
{
  int e = blockIdx.x * 256 + threadIdx.x;
  if (e >= E) return;
  int d = dst[e];
#pragma unroll
  for (int h = 0; h < 3; ++h) {
    float mv = fdec(menc[d*3+h]);
    float ex = __expf(tb[(size_t)e*3+h] - mv);
    tb[(size_t)e*3+h] = ex;
    atomicAdd(&ssum[d*3+h], ex);
  }
}

__global__ __launch_bounds__(256)
void att_scatter(const float* __restrict__ tb, const float* __restrict__ ssum,
                 const float* __restrict__ hbuf, const int* __restrict__ src,
                 const int* __restrict__ dst, float* __restrict__ nout, int E)
{
  long long gid = (long long)blockIdx.x * 256 + threadIdx.x;
  if (gid >= (long long)E * 64) return;
  int e = (int)(gid >> 6), dch = (int)(gid & 63);
  int s = src[e], d = dst[e];
#pragma unroll
  for (int h = 0; h < 3; ++h) {
    float alpha = tb[(size_t)e*3+h] / (ssum[d*3+h] + 1e-16f);
    atomicAdd(&nout[(size_t)d*HD + h*64 + dch],
              alpha * hbuf[(size_t)s*HD + h*64 + dch]);
  }
}

__global__ __launch_bounds__(256)
void node_mean(const float* __restrict__ nout, float* __restrict__ xo, int Nn)
{
  long long gid = (long long)blockIdx.x * 256 + threadIdx.x;
  if (gid >= (long long)Nn * 64) return;
  int n = (int)(gid >> 6), d = (int)(gid & 63);
  xo[gid] = (nout[(size_t)n*HD + d] + nout[(size_t)n*HD + 64 + d] +
             nout[(size_t)n*HD + 128 + d]) * (1.f / 3.f);
}

// ---------- batch norm (deterministic two-pass) ----------
__global__ __launch_bounds__(256)
void bn_part(const float* __restrict__ X, float* __restrict__ part,
             int rows, int chunk, int relu)
{
  __shared__ float ls[256], lq[256];
  int t = threadIdx.x, col = t & 63, grp = t >> 6;
  int r0 = blockIdx.x * chunk;
  int r1 = min(rows, r0 + chunk);
  float s = 0.f, q = 0.f;
  for (int r = r0 + grp; r < r1; r += 4) {
    float v = X[(size_t)r * 64 + col];
    if (relu) v = fmaxf(v, 0.f);
    s += v; q += v * v;
  }
  ls[t] = s; lq[t] = q;
  __syncthreads();
  if (t < 64) {
    s = ls[t] + ls[t+64] + ls[t+128] + ls[t+192];
    q = lq[t] + lq[t+64] + lq[t+128] + lq[t+192];
    part[blockIdx.x * 128 + t]      = s;
    part[blockIdx.x * 128 + 64 + t] = q;
  }
}

__global__ __launch_bounds__(64)
void bn_final(const float* __restrict__ part, const float* __restrict__ g,
              const float* __restrict__ b, float* __restrict__ coef,
              int rows, int nblk)
{
  int f = threadIdx.x;
  float s = 0.f, q = 0.f;
  for (int bb = 0; bb < nblk; ++bb) { s += part[bb*128+f]; q += part[bb*128+64+f]; }
  float mu  = s / rows;
  float var = fmaxf(q / rows - mu * mu, 0.f);
  float sc  = g[f] * rsqrtf(var + 1e-5f);
  coef[f]      = sc;
  coef[64 + f] = b[f] - mu * sc;
}

__global__ __launch_bounds__(256)
void bn_apply(const float* __restrict__ X, const float* __restrict__ coef,
              float* __restrict__ Y, int rows, int relu)
{
  long long gid = (long long)blockIdx.x * 256 + threadIdx.x;
  if (gid >= (long long)rows * 64) return;
  int f = (int)(gid & 63);
  float v = X[gid];
  if (relu) v = fmaxf(v, 0.f);
  Y[gid] = v * coef[f] + coef[64 + f];
}

// ---------- pooling + head ----------
__global__ __launch_bounds__(256)
void pool_add(const float* __restrict__ X, const int* __restrict__ batch,
              float* __restrict__ g, int rows)
{
  long long gid = (long long)blockIdx.x * 256 + threadIdx.x;
  if (gid >= (long long)rows * 64) return;
  int n = (int)(gid >> 6), d = (int)(gid & 63);
  atomicAdd(&g[(size_t)batch[n] * 64 + d], fmaxf(X[gid], 0.f));
}

__global__ __launch_bounds__(64)
void bn_small(const float* __restrict__ X, const float* __restrict__ g,
              const float* __restrict__ b, float* __restrict__ Y, int rows)
{
  int f = threadIdx.x;
  float s = 0.f, q = 0.f;
  for (int r = 0; r < rows; ++r) { float v = X[r*64+f]; s += v; q += v*v; }
  float mu  = s / rows;
  float var = fmaxf(q / rows - mu * mu, 0.f);
  float sc  = g[f] * rsqrtf(var + 1e-5f);
  float sh  = b[f] - mu * sc;
  for (int r = 0; r < rows; ++r) Y[r*64+f] = X[r*64+f] * sc + sh;
}

__global__ __launch_bounds__(256)
void lin_relu(const float* __restrict__ X, const float* __restrict__ Wl,
              const float* __restrict__ bl, float* __restrict__ Y)
{
  int gid = blockIdx.x * 256 + threadIdx.x;   // N_G*64 = 8192
  int r = gid >> 6, c = gid & 63;
  float a = bl[c];
  for (int k = 0; k < 64; ++k) a += X[r*64+k] * Wl[k*64+c];
  Y[gid] = fmaxf(a, 0.f);
}

__global__ __launch_bounds__(32)
void cls_logsoftmax(const float* __restrict__ X, const float* __restrict__ Wc,
                    const float* __restrict__ bc, float* __restrict__ out)
{
  __shared__ float lg[N_C];
  __shared__ float lse;
  int r = blockIdx.x, c = threadIdx.x;
  if (c < N_C) {
    float a = bc[c];
    for (int k = 0; k < 64; ++k) a += X[r*64+k] * Wc[k*N_C+c];
    lg[c] = a;
  }
  __syncthreads();
  if (c == 0) {
    float m = lg[0];
    for (int i = 1; i < N_C; ++i) m = fmaxf(m, lg[i]);
    float s = 0.f;
    for (int i = 0; i < N_C; ++i) s += __expf(lg[i] - m);
    lse = m + __logf(s);
  }
  __syncthreads();
  if (c < N_C) out[r*N_C+c] = lg[c] - lse;
}

// ---------- host orchestration ----------
extern "C" void kernel_launch(void* const* d_in, const int* in_sizes, int n_in,
                              void* d_out, int out_size, void* d_ws, size_t ws_size,
                              hipStream_t stream)
{
  const int N = N_NODES, E = N_EDGES;
  const float* x         = (const float*)d_in[0];
  const float* edge_attr = (const float*)d_in[1];
  const int*   src       = (const int*)d_in[2];
  const int*   dst       = src + E;
  const int*   batch     = (const int*)d_in[3];
  const float* bn_feat_g = (const float*)d_in[4];
  const float* bn_feat_b = (const float*)d_in[5];
  const float* fWn0      = (const float*)d_in[6];
  const float* fWe0      = (const float*)d_in[7];
  const float* fWn_r     = (const float*)d_in[8];
  const float* fWe_r     = (const float*)d_in[9];
  const float* fa_s      = (const float*)d_in[10];
  const float* fa_d      = (const float*)d_in[11];
  const float* fa_e      = (const float*)d_in[12];
  const float* cWn       = (const float*)d_in[13];
  const float* cWe       = (const float*)d_in[14];
  const float* ca_s      = (const float*)d_in[15];
  const float* ca_d      = (const float*)d_in[16];
  const float* ca_e      = (const float*)d_in[17];
  const float* bns_g     = (const float*)d_in[18];
  const float* bns_b     = (const float*)d_in[19];
  const float* bn_fc_g   = (const float*)d_in[20];
  const float* bn_fc_b   = (const float*)d_in[21];
  const float* lin_W     = (const float*)d_in[22];
  const float* lin_b     = (const float*)d_in[23];
  const float* bn_hid_g  = (const float*)d_in[24];
  const float* bn_hid_b  = (const float*)d_in[25];
  const float* cls_W     = (const float*)d_in[26];
  const float* cls_b     = (const float*)d_in[27];

  float* ws = (float*)d_ws;
  size_t off = 0;
  auto take = [&](size_t n) { float* p = ws + off; off += n; return p; };
  float*    xn   = take((size_t)N * 64);
  float*    xa   = take((size_t)N * 64);
  float*    xb   = take((size_t)N * 64);
  float*    hb   = take((size_t)N * HD);
  float*    nout = take((size_t)N * HD);
  float*    dsb  = take((size_t)N * 3);
  float*    ddb  = take((size_t)N * 3);
  unsigned* menc = (unsigned*)take((size_t)N * 3);
  float*    ssum = take((size_t)N * 3);
  float*    eF   = take((size_t)E * 64);
  float*    eA   = take((size_t)E * 64);
  float*    eB   = take((size_t)E * 64);
  float*    tb   = take((size_t)E * 3);
  float*    deb  = take((size_t)E * 3);
  float*    part = take((size_t)BN_BLOCKS * 128);
  float*    coef = take(128);
  float*    gp   = take((size_t)N_G * 64);
  float*    g2   = take((size_t)N_G * 64);
  float*    g3   = take((size_t)N_G * 64);
  float*    g4   = take((size_t)N_G * 64);

  auto zero = [&](float* p, long long n) {
    fill_zero<<<(unsigned)((n + 255) / 256), 256, 0, stream>>>(p, n);
  };
  auto bn = [&](const float* Xp, const float* gg, const float* bb, float* Yp, int relu) {
    int chunk = (N + BN_BLOCKS - 1) / BN_BLOCKS;
    bn_part<<<BN_BLOCKS, 256, 0, stream>>>(Xp, part, N, chunk, relu);
    bn_final<<<1, 64, 0, stream>>>(part, gg, bb, coef, N, BN_BLOCKS);
    long long tot = (long long)N * 64;
    bn_apply<<<(unsigned)((tot + 255) / 256), 256, 0, stream>>>(Xp, coef, Yp, N, relu);
  };
  auto layer = [&](const float* xin, int rn, const float* ein, int Kin, int re,
                   const float* Wn, const float* We, const float* as_,
                   const float* ad_, const float* ae_, float* xo, float* eo) {
    gemm_att<0><<<(N + 63) / 64, 128, 0, stream>>>(xin, Wn, as_, ad_, hb, dsb, ddb, N, 64, rn);
    gemm_att<1><<<(E + 63) / 64, 128, 0, stream>>>(ein, We, ae_, nullptr, nullptr, eo, deb, E, Kin, re);
    zero((float*)menc, (long long)N * 3);
    zero(ssum, (long long)N * 3);
    zero(nout, (long long)N * HD);
    att_logits_max<<<(E + 255) / 256, 256, 0, stream>>>(dsb, ddb, deb, src, dst, tb, menc, E);
    att_expsum<<<(E + 255) / 256, 256, 0, stream>>>(tb, dst, menc, ssum, E);
    long long te = (long long)E * 64;
    att_scatter<<<(unsigned)((te + 255) / 256), 256, 0, stream>>>(tb, ssum, hb, src, dst, nout, E);
    long long tn = (long long)N * 64;
    node_mean<<<(unsigned)((tn + 255) / 256), 256, 0, stream>>>(nout, xo, N);
  };

  // feature BN
  bn(x, bn_feat_g, bn_feat_b, xn, 0);

  // first EGAT block (ReLU on inputs of layers 1,2; consumers apply ReLU to outputs)
  layer(xn, 0, edge_attr, 8, 0, fWn0, fWe0, fa_s, fa_d, fa_e, xa, eA);
  layer(xa, 1, eA, 64, 1, fWn_r, fWe_r, fa_s + HD, fa_d + HD, fa_e + HD, xb, eB);
  layer(xb, 1, eB, 64, 1, fWn_r + 64 * HD, fWe_r + 64 * HD,
        fa_s + 2 * HD, fa_d + 2 * HD, fa_e + 2 * HD, xa, eF);

  float* cur = xa;
  float* alt = xb;
  for (int i = 0; i < 3; ++i) {
    bn(cur, bns_g + i * 64, bns_b + i * 64, xn, 1);   // relu(x) then BN
    const float* Wn0 = cWn + (size_t)(i * 3 + 0) * 64 * HD;
    const float* Wn1 = cWn + (size_t)(i * 3 + 1) * 64 * HD;
    const float* Wn2 = cWn + (size_t)(i * 3 + 2) * 64 * HD;
    const float* We0 = cWe + (size_t)(i * 3 + 0) * 64 * HD;
    const float* We1 = cWe + (size_t)(i * 3 + 1) * 64 * HD;
    const float* We2 = cWe + (size_t)(i * 3 + 2) * 64 * HD;
    const float* s0 = ca_s + (size_t)(i * 3 + 0) * HD;
    const float* s1 = ca_s + (size_t)(i * 3 + 1) * HD;
    const float* s2 = ca_s + (size_t)(i * 3 + 2) * HD;
    const float* d0 = ca_d + (size_t)(i * 3 + 0) * HD;
    const float* d1 = ca_d + (size_t)(i * 3 + 1) * HD;
    const float* d2 = ca_d + (size_t)(i * 3 + 2) * HD;
    const float* e0 = ca_e + (size_t)(i * 3 + 0) * HD;
    const float* e1 = ca_e + (size_t)(i * 3 + 1) * HD;
    const float* e2 = ca_e + (size_t)(i * 3 + 2) * HD;
    layer(xn,  0, eF, 64, 1, Wn0, We0, s0, d0, e0, alt, eA);
    layer(alt, 1, eA, 64, 1, Wn1, We1, s1, d1, e1, cur, eB);
    layer(cur, 1, eB, 64, 1, Wn2, We2, s2, d2, e2, alt, eA);
    float* t = cur; cur = alt; alt = t;
  }

  // global_add_pool of relu(x)
  zero(gp, (long long)N_G * 64);
  long long tn = (long long)N * 64;
  pool_add<<<(unsigned)((tn + 255) / 256), 256, 0, stream>>>(cur, batch, gp, N);

  // head: BN -> relu(lin) -> BN -> cls + log_softmax
  bn_small<<<1, 64, 0, stream>>>(gp, bn_fc_g, bn_fc_b, g2, N_G);
  lin_relu<<<(N_G * 64) / 256, 256, 0, stream>>>(g2, lin_W, lin_b, g3);
  bn_small<<<1, 64, 0, stream>>>(g3, bn_hid_g, bn_hid_b, g4, N_G);
  cls_logsoftmax<<<N_G, 32, 0, stream>>>(g4, cls_W, cls_b, (float*)d_out);

  (void)in_sizes; (void)n_in; (void)out_size; (void)ws_size;
}
